// GNN_28089086116705
// MI455X (gfx1250) — matrix operand order repaired and checked by reading, not compiled, not run on hardware
//
#include <hip/hip_runtime.h>
#include <hip/hip_bf16.h>

// CDNA5 (gfx1250) GNN pipeline: RGCN + MFConv blocks with bf16 WMMA GEMMs.
// wave32; v_wmma_f32_16x16x32_bf16 for all dense [*,64]x[64,64] transforms.

typedef __attribute__((ext_vector_type(16))) __bf16 v16bf;
typedef __attribute__((ext_vector_type(8)))  float  v8f;

#define NHID 64
#define NREL 4
#define MAXDEG 10
#define NGRAPH 16

// ---------------------------------------------------------------------------
// WMMA fragment helpers (layouts per CDNA5 ISA 7.12.2, wave32)
// ---------------------------------------------------------------------------
__device__ __forceinline__ v16bf load_a_frag(const __bf16* row, int kbase) {
  // A 16x32 bf16: lane holds row m=lane%16; elements 0..15 = K ks*32+(lane/16)*16+0..15
  // (kbase = ks*32 + (lane/16)*8, so start = kbase + (kbase & 8)).
  v16bf a;
  const int k0 = kbase + (kbase & 8);
#pragma unroll
  for (int j = 0; j < 16; ++j) {
    a[j] = row[k0 + j];
  }
  return a;
}

__device__ __forceinline__ v16bf load_b_frag(const __bf16* wt, int n, int kb) {
  // B 32x16 bf16 from transposed weights Wt[n][k]: lane holds col n=lane%16,
  // 16 contiguous K values starting at kb = ks*32 + (lane/16)*16.
  const __bf16* p = wt + ((size_t)n << 6) + kb;
  v16bf b;
#pragma unroll
  for (int j = 0; j < 16; ++j) b[j] = p[j];
  return b;
}

__device__ __forceinline__ v8f wmma_bf16(v16bf a, v16bf b, v8f c) {
  return __builtin_amdgcn_wmma_f32_16x16x32_bf16(
      /*neg_a=*/false, a, /*neg_b=*/false, b,
      /*c_mod=*/(short)0, c, /*reuse_a=*/false, /*reuse_b=*/false);
}

// ---------------------------------------------------------------------------
// Preprocessing kernels
// ---------------------------------------------------------------------------
__global__ void etype_kernel(const float* __restrict__ eattr,
                             int* __restrict__ etype, int E) {
  int e = blockIdx.x * blockDim.x + threadIdx.x;
  if (e >= E) return;
  const float4 a = *reinterpret_cast<const float4*>(eattr + (size_t)e * 4);
  int bi = 0; float bv = a.x;
  if (a.y > bv) { bv = a.y; bi = 1; }
  if (a.z > bv) { bv = a.z; bi = 2; }
  if (a.w > bv) { bv = a.w; bi = 3; }
  etype[e] = bi;
}

// Convert all weight matrices to bf16, transposed: Wt[o*64 + d] = W[d*64 + o].
// Matrix layout in wbf: [0..9]  rgcn: blk*5 + r (r=0..3), root at blk*5+4
//                       [10..31] mf_wl: 10 + blk*11 + k
//                       [32..53] mf_wr: 32 + blk*11 + k
__global__ void prep_weights(const float* __restrict__ rgcn_w,
                             const float* __restrict__ rgcn_root,
                             const float* __restrict__ mf_wl,
                             const float* __restrict__ mf_wr,
                             __bf16* __restrict__ wbf) {
  int t = blockIdx.x * blockDim.x + threadIdx.x;
  if (t >= 54 * 4096) return;
  int mIdx = t >> 12;
  int j = t & 4095;
  int o = j >> 6, d = j & 63;
  const float* src;
  if (mIdx < 10) {
    int blk = mIdx / 5, sub = mIdx % 5;
    src = (sub < 4) ? (rgcn_w + ((size_t)(blk * 4 + sub) << 12))
                    : (rgcn_root + ((size_t)blk << 12));
  } else if (mIdx < 32) {
    src = mf_wl + ((size_t)(mIdx - 10) << 12);
  } else {
    src = mf_wr + ((size_t)(mIdx - 32) << 12);
  }
  wbf[((size_t)mIdx << 12) + j] = (__bf16)src[d * 64 + o];
}

__global__ void embed_kernel(const float* __restrict__ x,
                             const float* __restrict__ Wemb,
                             const float* __restrict__ bemb,
                             float* __restrict__ h, int N) {
  long long t = (long long)blockIdx.x * blockDim.x + threadIdx.x;
  if (t >= (long long)N * NHID) return;
  int n = (int)(t >> 6), o = (int)(t & 63);
  float acc = bemb[o];
#pragma unroll
  for (int k = 0; k < 5; ++k) acc += x[(size_t)n * 5 + k] * Wemb[k * NHID + o];
  h[t] = acc;
}

// ---------------------------------------------------------------------------
// Edge aggregation (16 threads/edge, float4 gather + f32 atomics; L2-resident)
// ---------------------------------------------------------------------------
__global__ void rgcn_agg(const float* __restrict__ h, const int* __restrict__ src,
                         const int* __restrict__ dst, const int* __restrict__ etype,
                         float* __restrict__ ssum, float* __restrict__ cnt,
                         int E, int N) {
  long long gid = (long long)blockIdx.x * blockDim.x + threadIdx.x;
  int e = (int)(gid >> 4);
  if (e >= E) return;
  int c4 = (int)(gid & 15) << 2;
  int s = src[e], d = dst[e], r = etype[e];
  const float4 v = *reinterpret_cast<const float4*>(h + ((size_t)s << 6) + c4);
  float* base = ssum + (((size_t)r * N + d) << 6) + c4;
  atomicAdd(base + 0, v.x);
  atomicAdd(base + 1, v.y);
  atomicAdd(base + 2, v.z);
  atomicAdd(base + 3, v.w);
  if (c4 == 0) atomicAdd(cnt + (size_t)r * N + d, 1.0f);
}

__global__ void mf_agg(const float* __restrict__ h, const int* __restrict__ src,
                       const int* __restrict__ dst, float* __restrict__ msum,
                       float* __restrict__ cnt2, int E) {
  long long gid = (long long)blockIdx.x * blockDim.x + threadIdx.x;
  int e = (int)(gid >> 4);
  if (e >= E) return;
  int c4 = (int)(gid & 15) << 2;
  int s = src[e], d = dst[e];
  const float4 v = *reinterpret_cast<const float4*>(h + ((size_t)s << 6) + c4);
  float* base = msum + ((size_t)d << 6) + c4;
  atomicAdd(base + 0, v.x);
  atomicAdd(base + 1, v.y);
  atomicAdd(base + 2, v.z);
  atomicAdd(base + 3, v.w);
  if (c4 == 0) atomicAdd(cnt2 + d, 1.0f);
}

// ---------------------------------------------------------------------------
// RGCN combine: out = sum_r mean_r @ W[r] + h @ root + b, ReLU.  WMMA bf16.
// 128 threads = 4 waves; wave w owns output cols [16w,16w+16); block owns a
// 16-node row tile. 5 matrices x K=64 -> 10 v_wmma per wave.
// ---------------------------------------------------------------------------
__global__ void __launch_bounds__(128) rgcn_combine(
    const float* __restrict__ ssum, const float* __restrict__ cnt,
    const float* __restrict__ hcur, const __bf16* __restrict__ wbase,
    const float* __restrict__ bias, float* __restrict__ hmid, int N) {
  __shared__ __bf16 sA[5][16][NHID];  // 4 relation means + self, bf16
  __shared__ float sInv[4][16];
  const int tid = threadIdx.x;
  const int lane = tid & 31, wv = tid >> 5;
  const int nLoc = lane & 15, khalf = lane >> 4;
  const int col = (wv << 4) + nLoc;
  const int tile = blockIdx.x;

  // Per-wave B fragments for 5 matrices x 2 k-steps (weights L2-resident).
  v16bf bfrag[5][2];
#pragma unroll
  for (int m = 0; m < 5; ++m)
#pragma unroll
    for (int ks = 0; ks < 2; ++ks)
      bfrag[m][ks] = load_b_frag(wbase + ((size_t)m << 12), col, ks * 32 + khalf * 16);

  if (tid < 64) {
    int r = tid >> 4, m = tid & 15;
    int node = tile * 16 + m;
    float c = (node < N) ? cnt[(size_t)r * N + node] : 1.0f;
    sInv[r][m] = 1.0f / fmaxf(c, 1.0f);
  }
  __syncthreads();

  for (int i = tid; i < 5 * 16 * NHID; i += 128) {
    int mat = i >> 10, rem = i & 1023;
    int m = rem >> 6, k = rem & 63;
    int node = tile * 16 + m;
    float v = 0.0f;
    if (node < N) {
      v = (mat < 4) ? ssum[(((size_t)mat * N + node) << 6) + k] * sInv[mat][m]
                    : hcur[((size_t)node << 6) + k];
    }
    sA[mat][m][k] = (__bf16)v;
  }
  __syncthreads();

  v8f acc = {};
#pragma unroll
  for (int mat = 0; mat < 5; ++mat) {
    const __bf16* row = &sA[mat][nLoc][0];
#pragma unroll
    for (int ks = 0; ks < 2; ++ks) {
      v16bf a = load_a_frag(row, ks * 32 + khalf * 8);
      acc = wmma_bf16(a, bfrag[mat][ks], acc);
    }
  }

  float bcol = bias[col];
#pragma unroll
  for (int j = 0; j < 8; ++j) {
    int m = j + (khalf << 3);
    int node = tile * 16 + m;
    if (node < N) {
      float v = acc[j] + bcol;
      hmid[((size_t)node << 6) + col] = fmaxf(v, 0.0f);
    }
  }
}

// ---------------------------------------------------------------------------
// Degree bucketing (so every 16-row WMMA tile shares one degree weight)
// ---------------------------------------------------------------------------
__global__ void deg_hist(const float* __restrict__ cnt2, int* __restrict__ deg,
                         int* __restrict__ bcnt, int N) {
  int n = blockIdx.x * blockDim.x + threadIdx.x;
  if (n >= N) return;
  int d = (int)cnt2[n];
  d = d < 0 ? 0 : (d > MAXDEG ? MAXDEG : d);
  deg[n] = d;
  atomicAdd(bcnt + d, 1);
}

__global__ void bucket_off(const int* __restrict__ bcnt, int* __restrict__ off,
                           int* __restrict__ cursor) {
  if (blockIdx.x == 0 && threadIdx.x == 0) {
    int o = 0;
    for (int k = 0; k <= MAXDEG; ++k) {
      off[k] = o;
      o += (bcnt[k] + 15) & ~15;  // 16-align each bucket
      cursor[k] = 0;
    }
    off[MAXDEG + 1] = o;
  }
}

__global__ void bucket_scatter(const int* __restrict__ deg, const int* __restrict__ off,
                               int* __restrict__ cursor, int* __restrict__ list, int N) {
  int n = blockIdx.x * blockDim.x + threadIdx.x;
  if (n >= N) return;
  int d = deg[n];
  int p = off[d] + atomicAdd(cursor + d, 1);
  list[p] = n;
}

// ---------------------------------------------------------------------------
// MF combine: out = mean @ Wl[deg] + bl[deg] + h @ Wr[deg], via bucketed list.
// 4 v_wmma per wave; padded rows (list==-1) are zero-filled and not stored.
// ---------------------------------------------------------------------------
__global__ void __launch_bounds__(128) mf_combine(
    const float* __restrict__ msum, const float* __restrict__ cnt2,
    const float* __restrict__ hmid, const __bf16* __restrict__ wlbase,
    const __bf16* __restrict__ wrbase, const float* __restrict__ blbase,
    const int* __restrict__ list, const int* __restrict__ off,
    float* __restrict__ hout, int applyRelu) {
  __shared__ __bf16 sA[2][16][NHID];
  __shared__ int sNode[16];
  __shared__ float sInv[16];
  const int tileStart = blockIdx.x << 4;
  if (tileStart >= off[MAXDEG + 1]) return;

  int kdeg = 0;
#pragma unroll
  for (int k = 0; k < MAXDEG; ++k)
    if (tileStart >= off[k + 1]) kdeg = k + 1;

  const int tid = threadIdx.x;
  if (tid < 16) {
    int node = list[tileStart + tid];
    sNode[tid] = node;
    float c = (node >= 0) ? cnt2[node] : 1.0f;
    sInv[tid] = 1.0f / fmaxf(c, 1.0f);
  }
  __syncthreads();

  for (int i = tid; i < 2 * 16 * NHID; i += 128) {
    int mat = i >> 10, rem = i & 1023;
    int m = rem >> 6, k = rem & 63;
    int node = sNode[m];
    float v = 0.0f;
    if (node >= 0) {
      v = (mat == 0) ? msum[((size_t)node << 6) + k] * sInv[m]
                     : hmid[((size_t)node << 6) + k];
    }
    sA[mat][m][k] = (__bf16)v;
  }
  __syncthreads();

  const int lane = tid & 31, wv = tid >> 5;
  const int nLoc = lane & 15, khalf = lane >> 4;
  const int col = (wv << 4) + nLoc;
  const __bf16* wl = wlbase + ((size_t)kdeg << 12);
  const __bf16* wr = wrbase + ((size_t)kdeg << 12);

  v8f acc = {};
#pragma unroll
  for (int ks = 0; ks < 2; ++ks) {
    v16bf a0 = load_a_frag(&sA[0][nLoc][0], ks * 32 + khalf * 8);
    v16bf b0 = load_b_frag(wl, col, ks * 32 + khalf * 16);
    acc = wmma_bf16(a0, b0, acc);
    v16bf a1 = load_a_frag(&sA[1][nLoc][0], ks * 32 + khalf * 8);
    v16bf b1 = load_b_frag(wr, col, ks * 32 + khalf * 16);
    acc = wmma_bf16(a1, b1, acc);
  }

  float bcol = blbase[(size_t)kdeg * NHID + col];
#pragma unroll
  for (int j = 0; j < 8; ++j) {
    int m = j + (khalf << 3);
    int node = sNode[m];
    if (node >= 0) {
      float v = acc[j] + bcol;
      if (applyRelu) v = fmaxf(v, 0.0f);
      hout[((size_t)node << 6) + col] = v;
    }
  }
}

// ---------------------------------------------------------------------------
// Pooling + head
// ---------------------------------------------------------------------------
__global__ void pool_kernel(const float* __restrict__ h, const int* __restrict__ batch,
                            float* __restrict__ pooled, float* __restrict__ gcnt, int N) {
  long long gid = (long long)blockIdx.x * blockDim.x + threadIdx.x;
  if (gid >= (long long)N * NHID) return;
  int n = (int)(gid >> 6), c = (int)(gid & 63);
  int g = batch[n];
  atomicAdd(pooled + (size_t)g * NHID + c, h[((size_t)n << 6) + c]);
  if (c == 0) atomicAdd(gcnt + g, 1.0f);
}

__global__ void head_kernel(const float* __restrict__ pooled, const float* __restrict__ gcnt,
                            const float* __restrict__ W1, const float* __restrict__ b1,
                            const float* __restrict__ W2, const float* __restrict__ b2,
                            float* __restrict__ out) {
  __shared__ float red[NHID];
  int o = threadIdx.x;  // 64 threads
  for (int g = 0; g < NGRAPH; ++g) {
    float inv = 1.0f / fmaxf(gcnt[g], 1.0f);
    float acc = b1[o];
    for (int c = 0; c < NHID; ++c)
      acc += pooled[(size_t)g * NHID + c] * inv * W1[c * NHID + o];
    red[o] = fmaxf(acc, 0.0f) * W2[o];
    __syncthreads();
    if (o == 0) {
      float s = 0.0f;
      for (int i = 0; i < NHID; ++i) s += red[i];
      out[g] = s + b2[0];
    }
    __syncthreads();
  }
}

// ---------------------------------------------------------------------------
// Host driver
// ---------------------------------------------------------------------------
extern "C" void kernel_launch(void* const* d_in, const int* in_sizes, int n_in,
                              void* d_out, int out_size, void* d_ws, size_t ws_size,
                              hipStream_t stream) {
  (void)n_in; (void)out_size; (void)ws_size;
  const float* x        = (const float*)d_in[0];
  const int*   ei       = (const int*)d_in[1];
  const float* eattr    = (const float*)d_in[2];
  const int*   batch    = (const int*)d_in[3];
  const float* Wemb     = (const float*)d_in[4];
  const float* bemb     = (const float*)d_in[5];
  const float* rgcn_w   = (const float*)d_in[6];
  const float* rgcn_root= (const float*)d_in[7];
  const float* rgcn_b   = (const float*)d_in[8];
  const float* mf_wl    = (const float*)d_in[9];
  const float* mf_bl    = (const float*)d_in[10];
  const float* mf_wr    = (const float*)d_in[11];
  const float* W1       = (const float*)d_in[12];
  const float* b1       = (const float*)d_in[13];
  const float* W2       = (const float*)d_in[14];
  const float* b2       = (const float*)d_in[15];
  float* out = (float*)d_out;

  const int N = in_sizes[0] / 5;
  const int E = in_sizes[1] / 2;
  const int* srcp = ei;
  const int* dstp = ei + E;

  char* ws = (char*)d_ws;
  size_t p = 0;
  auto carve = [&](size_t bytes) {
    size_t o = p;
    p += (bytes + 255) & ~(size_t)255;
    return o;
  };
  int*    etype = (int*)   (ws + carve((size_t)E * 4));
  float*  hcur  = (float*) (ws + carve((size_t)N * NHID * 4));
  float*  hmid  = (float*) (ws + carve((size_t)N * NHID * 4));
  float*  hnext = (float*) (ws + carve((size_t)N * NHID * 4));
  float*  ssum  = (float*) (ws + carve((size_t)NREL * N * NHID * 4));
  float*  cnt   = (float*) (ws + carve((size_t)NREL * N * 4));
  float*  msum  = (float*) (ws + carve((size_t)N * NHID * 4));
  float*  cnt2  = (float*) (ws + carve((size_t)N * 4));
  int*    deg   = (int*)   (ws + carve((size_t)N * 4));
  __bf16* wbf   = (__bf16*)(ws + carve((size_t)54 * 4096 * 2));
  int*    bcnt  = (int*)   (ws + carve(11 * 4));
  int*    curs  = (int*)   (ws + carve(11 * 4));
  int*    offA  = (int*)   (ws + carve(12 * 4));
  int*    list  = (int*)   (ws + carve((size_t)(N + 11 * 16) * 4));
  float*  pooled= (float*) (ws + carve((size_t)NGRAPH * NHID * 4));
  float*  gcnt  = (float*) (ws + carve(NGRAPH * 4));

  const int numTiles   = (N + 15) / 16;
  const int maxMfTiles = (N + 11 * 16) / 16;
  const long long aggThreads = (long long)E * 16;
  const int aggBlocks = (int)((aggThreads + 255) / 256);
  const int nodeElemBlocks = (int)(((long long)N * NHID + 255) / 256);

  etype_kernel<<<(E + 255) / 256, 256, 0, stream>>>(eattr, etype, E);
  prep_weights<<<(54 * 4096 + 255) / 256, 256, 0, stream>>>(rgcn_w, rgcn_root, mf_wl, mf_wr, wbf);
  embed_kernel<<<nodeElemBlocks, 256, 0, stream>>>(x, Wemb, bemb, hcur, N);

  for (int blk = 0; blk < 2; ++blk) {
    // --- RGCN ---
    hipMemsetAsync(ssum, 0, (size_t)NREL * N * NHID * 4, stream);
    hipMemsetAsync(cnt, 0, (size_t)NREL * N * 4, stream);
    rgcn_agg<<<aggBlocks, 256, 0, stream>>>(hcur, srcp, dstp, etype, ssum, cnt, E, N);
    rgcn_combine<<<numTiles, 128, 0, stream>>>(
        ssum, cnt, hcur, wbf + (size_t)blk * 5 * 4096, rgcn_b + blk * NHID, hmid, N);

    // --- MFConv ---
    hipMemsetAsync(msum, 0, (size_t)N * NHID * 4, stream);
    hipMemsetAsync(cnt2, 0, (size_t)N * 4, stream);
    mf_agg<<<aggBlocks, 256, 0, stream>>>(hmid, srcp, dstp, msum, cnt2, E);

    hipMemsetAsync(bcnt, 0, 11 * 4, stream);
    hipMemsetAsync(list, 0xFF, (size_t)(N + 11 * 16) * 4, stream);  // -1 fill
    deg_hist<<<(N + 255) / 256, 256, 0, stream>>>(cnt2, deg, bcnt, N);
    bucket_off<<<1, 1, 0, stream>>>(bcnt, offA, curs);
    bucket_scatter<<<(N + 255) / 256, 256, 0, stream>>>(deg, offA, curs, list, N);

    float* hout = (blk == 0) ? hcur : hnext;  // fold inter-block ReLU into blk0 output
    mf_combine<<<maxMfTiles, 128, 0, stream>>>(
        msum, cnt2, hmid,
        wbf + (size_t)(10 + blk * 11) * 4096,
        wbf + (size_t)(32 + blk * 11) * 4096,
        mf_bl + (size_t)blk * (MAXDEG + 1) * NHID,
        list, offA, hout, blk == 0 ? 1 : 0);
  }

  hipMemsetAsync(pooled, 0, (size_t)NGRAPH * NHID * 4, stream);
  hipMemsetAsync(gcnt, 0, NGRAPH * 4, stream);
  pool_kernel<<<nodeElemBlocks, 256, 0, stream>>>(hnext, batch, pooled, gcnt, N);
  head_kernel<<<1, 64, 0, stream>>>(pooled, gcnt, W1, b1, W2, b2, out);
}
